// PolymorphicDense_44306882626005
// MI455X (gfx1250) — matrix operand order repaired and checked by reading, not compiled, not run on hardware
//
#include <hip/hip_runtime.h>

#define B_SZ 8192
#define D_SZ 512
#define U_SZ 512
#define M_SZ 16
#define K_SZ 64

typedef __bf16 bf16_t;
typedef bf16_t v16bf __attribute__((ext_vector_type(16)));
typedef float  v8f   __attribute__((ext_vector_type(8)));

// round-to-nearest-even f32 -> bf16 (bit pattern)
static __device__ __forceinline__ unsigned short f32_to_bf16_rne(float f) {
    unsigned u = __float_as_uint(f);
    unsigned r = 0x7FFFu + ((u >> 16) & 1u);
    u += r;
    return (unsigned short)(u >> 16);
}

// ---------------------------------------------------------------------------
// Pack x [B,D] f32 into per-lane WMMA A-fragments (bf16 hi/lo split).
// Layout: frag = bTile*16 + dTile ; element addr = (frag*32 + lane)*16 + e
// A order used here (contiguous k, not the ISA table): lane<16 row=lane, e0..15 -> K=e ;
// lane>=16: +16 on K (matches B packing).
// ---------------------------------------------------------------------------
__global__ __launch_bounds__(256) void pack_x_kernel(const float* __restrict__ x,
                                                     unsigned short* __restrict__ xhi,
                                                     unsigned short* __restrict__ xlo) {
    int gid  = blockIdx.x * 256 + threadIdx.x;   // frag*32 + lane
    int lane = gid & 31;
    int frag = gid >> 5;
    int dTile = frag & 15;       // D/32 = 16
    int bTile = frag >> 4;
    int row   = bTile * 16 + (lane & 15);
    const float* src = x + (size_t)row * D_SZ + dTile * 32;
    int kadd = (lane >= 16) ? 16 : 0;

    unsigned short hi[16] __attribute__((aligned(16)));
    unsigned short lo[16] __attribute__((aligned(16)));
#pragma unroll
    for (int e = 0; e < 16; ++e) {
        int K = e + kadd;
        float f = src[K];
        unsigned short h = f32_to_bf16_rne(f);
        float hf = __uint_as_float((unsigned)h << 16);
        hi[e] = h;
        lo[e] = f32_to_bf16_rne(f - hf);
    }
    size_t o = (size_t)gid * 16;
    ((uint4*)(xhi + o))[0] = ((const uint4*)hi)[0];
    ((uint4*)(xhi + o))[1] = ((const uint4*)hi)[1];
    ((uint4*)(xlo + o))[0] = ((const uint4*)lo)[0];
    ((uint4*)(xlo + o))[1] = ((const uint4*)lo)[1];
}

// ---------------------------------------------------------------------------
// Pack kernels [M,D,U] f32 into per-lane WMMA B-fragments (bf16 hi/lo).
// Layout: frag = uTile*256 + ktg  (ktg = m*16 + dTile) ;
// B order used here (contiguous k, not the ISA table): lanes 0-15 col=lane hold K=0..15 (e=K), lanes 16-31 K=16..31.
// ---------------------------------------------------------------------------
__global__ __launch_bounds__(256) void pack_w_kernel(const float* __restrict__ w,
                                                     unsigned short* __restrict__ whi,
                                                     unsigned short* __restrict__ wlo) {
    int gid  = blockIdx.x * 256 + threadIdx.x;   // frag*32 + lane
    int lane = gid & 31;
    int frag = gid >> 5;
    int ktg   = frag & 255;      // M*D/32 = 256
    int uTile = frag >> 8;
    int m     = ktg >> 4;
    int dbase = (ktg & 15) * 32;
    int u     = uTile * 16 + (lane & 15);
    int kofs  = (lane >= 16) ? 16 : 0;

    unsigned short hi[16] __attribute__((aligned(16)));
    unsigned short lo[16] __attribute__((aligned(16)));
#pragma unroll
    for (int e = 0; e < 16; ++e) {
        float f = w[((size_t)m * D_SZ + dbase + kofs + e) * U_SZ + u];
        unsigned short h = f32_to_bf16_rne(f);
        float hf = __uint_as_float((unsigned)h << 16);
        hi[e] = h;
        lo[e] = f32_to_bf16_rne(f - hf);
    }
    size_t o = (size_t)gid * 16;
    ((uint4*)(whi + o))[0] = ((const uint4*)hi)[0];
    ((uint4*)(whi + o))[1] = ((const uint4*)hi)[1];
    ((uint4*)(wlo + o))[0] = ((const uint4*)lo)[0];
    ((uint4*)(wlo + o))[1] = ((const uint4*)lo)[1];
}

// ---------------------------------------------------------------------------
// sim[b,m] = 1/(||x[b]@key_kernel + key_bias - keys_map[m]|| + 1)
// one 64-thread block per batch row; thread t owns keyv[t].
// ---------------------------------------------------------------------------
__global__ __launch_bounds__(64) void sim_kernel(const float* __restrict__ x,
                                                 const float* __restrict__ key_kernel,
                                                 const float* __restrict__ key_bias,
                                                 const float* __restrict__ keys_map,
                                                 float* __restrict__ sim) {
    __shared__ float keyv[K_SZ];
    int b = blockIdx.x;
    int t = threadIdx.x;
    const float* xr = x + (size_t)b * D_SZ;
    float acc = key_bias[t];
    for (int d = 0; d < D_SZ; ++d)
        acc = fmaf(xr[d], key_kernel[(size_t)d * K_SZ + t], acc);
    keyv[t] = acc;
    __syncthreads();
    if (t < M_SZ) {
        float s = 0.0f;
#pragma unroll 8
        for (int k = 0; k < K_SZ; ++k) {
            float diff = keyv[k] - keys_map[(size_t)t * K_SZ + k];
            s = fmaf(diff, diff, s);
        }
        sim[(size_t)b * M_SZ + t] = 1.0f / (sqrtf(s) + 1.0f);
    }
}

// ---------------------------------------------------------------------------
// Main GEMM: per wave a 16(B) x 64(U) strip. For each mode m accumulate the
// bf16(hi/lo)-split GEMM partial in f32, then fold sim[b,m]/M and bias_m.
// ---------------------------------------------------------------------------
__global__ __launch_bounds__(256) void gemm_kernel(const unsigned short* __restrict__ xhi_,
                                                   const unsigned short* __restrict__ xlo_,
                                                   const unsigned short* __restrict__ whi_,
                                                   const unsigned short* __restrict__ wlo_,
                                                   const float* __restrict__ sim,
                                                   const float* __restrict__ biases,
                                                   float* __restrict__ out) {
    const v16bf* xhi = (const v16bf*)xhi_;
    const v16bf* xlo = (const v16bf*)xlo_;
    const v16bf* whi = (const v16bf*)whi_;
    const v16bf* wlo = (const v16bf*)wlo_;

    int lane   = threadIdx.x & 31;
    int wave   = threadIdx.x >> 5;
    int bTile  = blockIdx.x * 8 + wave;   // B/16 = 512 tiles
    int uTile0 = blockIdx.y * 4;          // U/16 = 32 tiles
    int half8  = (lane >= 16) ? 8 : 0;
    int bRow0  = bTile * 16 + half8;
    int ucol   = lane & 15;

    // A fragments for this wave's 16 rows, all 16 D-tiles, kept in registers
    v16bf aHi[16], aLo[16];
#pragma unroll
    for (int kt = 0; kt < 16; ++kt) {
        size_t aIdx = (size_t)(bTile * 16 + kt) * 32 + lane;
        aHi[kt] = xhi[aIdx];
        aLo[kt] = xlo[aIdx];
    }

    const v8f vzero = {0.f, 0.f, 0.f, 0.f, 0.f, 0.f, 0.f, 0.f};
    v8f accF[4] = {vzero, vzero, vzero, vzero};

    for (int m = 0; m < M_SZ; ++m) {
        v8f accM[4] = {vzero, vzero, vzero, vzero};
#pragma unroll
        for (int kt = 0; kt < 16; ++kt) {
            int ktg = m * 16 + kt;
#pragma unroll
            for (int j = 0; j < 4; ++j) {
                size_t bIdx = (size_t)((uTile0 + j) * 256 + ktg) * 32 + lane;
                v16bf bHi = whi[bIdx];
                v16bf bLo = wlo[bIdx];
                accM[j] = __builtin_amdgcn_wmma_f32_16x16x32_bf16(
                    false, aHi[kt], false, bHi, (short)0, accM[j], false, false);
                accM[j] = __builtin_amdgcn_wmma_f32_16x16x32_bf16(
                    false, aHi[kt], false, bLo, (short)0, accM[j], false, false);
                accM[j] = __builtin_amdgcn_wmma_f32_16x16x32_bf16(
                    false, aLo[kt], false, bHi, (short)0, accM[j], false, false);
            }
        }
        float biasv[4];
#pragma unroll
        for (int j = 0; j < 4; ++j)
            biasv[j] = biases[(size_t)m * U_SZ + (uTile0 + j) * 16 + ucol];
#pragma unroll
        for (int r = 0; r < 8; ++r) {
            float s = sim[(size_t)(bRow0 + r) * M_SZ + m] * (1.0f / (float)M_SZ);
#pragma unroll
            for (int j = 0; j < 4; ++j)
                accF[j][r] += s * (accM[j][r] + biasv[j]);
        }
    }

#pragma unroll
    for (int j = 0; j < 4; ++j)
#pragma unroll
        for (int r = 0; r < 8; ++r)
            out[(size_t)(bRow0 + r) * U_SZ + (uTile0 + j) * 16 + ucol] = accF[j][r];
}

extern "C" void kernel_launch(void* const* d_in, const int* in_sizes, int n_in,
                              void* d_out, int out_size, void* d_ws, size_t ws_size,
                              hipStream_t stream) {
    const float* x          = (const float*)d_in[0];
    const float* key_kernel = (const float*)d_in[1];
    const float* key_bias   = (const float*)d_in[2];
    const float* keys_map   = (const float*)d_in[3];
    const float* kernels    = (const float*)d_in[4];
    const float* biases     = (const float*)d_in[5];
    float* out = (float*)d_out;

    const size_t nx = (size_t)B_SZ * D_SZ;          // 4,194,304 elems
    const size_t nw = (size_t)M_SZ * D_SZ * U_SZ;   // 4,194,304 elems

    unsigned short* xhi = (unsigned short*)d_ws;
    unsigned short* xlo = xhi + nx;
    unsigned short* whi = xlo + nx;
    unsigned short* wlo = whi + nw;
    float*          sim = (float*)(wlo + nw);       // B*M f32 (512 KB)

    // pack x: (B/16)*(D/32) frags * 32 lanes = 262144 threads
    pack_x_kernel<<<(B_SZ / 16) * (D_SZ / 32) * 32 / 256, 256, 0, stream>>>(x, xhi, xlo);
    // pack w: (U/16)*(M*D/32) frags * 32 lanes = 262144 threads
    pack_w_kernel<<<(U_SZ / 16) * (M_SZ * D_SZ / 32) * 32 / 256, 256, 0, stream>>>(kernels, whi, wlo);
    // sim: one 64-thread block per batch row
    sim_kernel<<<B_SZ, 64, 0, stream>>>(x, key_kernel, key_bias, keys_map, sim);
    // main WMMA GEMM: 64 x 8 blocks of 8 waves
    gemm_kernel<<<dim3(B_SZ / 128, U_SZ / 64), 256, 0, stream>>>(xhi, xlo, whi, wlo,
                                                                 sim, biases, out);
}
